// PQCHead_21294447853615
// MI455X (gfx1250) — compile-verified, matrix operand order checked by reading, not run on hardware
//
#include <hip/hip_runtime.h>
#include <hip/hip_bf16.h>

typedef __attribute__((ext_vector_type(16))) _Float16 v16h;
typedef __attribute__((ext_vector_type(8)))  _Float16 v8h;
typedef __attribute__((ext_vector_type(8)))  float    v8f;

#define IN_F 2048
#define NQ 8
#define DIM 256
#define NCLS 1000
#define NCLS_PAD 1008
#define BATCH 8192

// workspace layout (byte offsets)
#define WS_UR 0
#define WS_UI (WS_UR + DIM*DIM*2)
#define WS_W2 (WS_UI + DIM*DIM*2)
#define WS_PW (WS_W2 + NCLS_PAD*DIM*2)

#define SIMK 32
#define SIM_BLOCKS 8

#define WAVES_PER_BLOCK 2
#define SMEM_PER_WAVE 17408   // X 8192B + P 8192B + cs 1024B

// Composite ring-CNOT permutation: new_state[j] = old_state[ring_perm(j)]
__device__ __forceinline__ int ring_perm(int j) {
  int y = j;
#pragma unroll
  for (int i = 7; i >= 0; --i) {        // apply pair (7,0) first ... (0,1) last
    int t = (i + 1) & 7;
    int cbit = (y >> (7 - i)) & 1;
    y ^= cbit << (7 - t);
  }
  return y;
}

// load 16 f16 A/B fragment elements: 8 at p, 8 at p+16 (matches 16-bit WMMA lane layout)
__device__ __forceinline__ v16h ldab(const _Float16* p) {
  v8h a0 = *(const v8h*)(p);
  v8h a1 = *(const v8h*)(p + 16);
  return __builtin_shufflevector(a0, a1, 0,1,2,3,4,5,6,7,8,9,10,11,12,13,14,15);
}

__global__ void pqc_precompute(const float* __restrict__ proj_w,
                               const float* __restrict__ q_weights,
                               const float* __restrict__ cls_w,
                               char* __restrict__ ws) {
  extern __shared__ char smem[];
  _Float16* Ur16 = (_Float16*)(ws + WS_UR);
  _Float16* Ui16 = (_Float16*)(ws + WS_UI);
  _Float16* W2   = (_Float16*)(ws + WS_W2);
  _Float16* PW   = (_Float16*)(ws + WS_PW);
  const int tid = threadIdx.x;
  const int bid = blockIdx.x;

  if (bid < SIM_BLOCKS) {
    // simulate SIMK columns of U in LDS: S[buf2][comp2][256][SIMK]
    float* S = (float*)smem;
    const int kbase = bid * SIMK;
    int cur = 0;
    for (int t = tid; t < DIM * SIMK; t += blockDim.x) {
      int j = t / SIMK, kk = t % SIMK;
      S[((cur*2 + 0)*DIM + j)*SIMK + kk] = (j == kbase + kk) ? 1.0f : 0.0f;
      S[((cur*2 + 1)*DIM + j)*SIMK + kk] = 0.0f;
    }
    __syncthreads();
    for (int l = 0; l < 4; ++l) {
      for (int w = 0; w < NQ; ++w) {
        const int bpos = 7 - w, mask = 1 << bpos;
        const float th = 0.5f * q_weights[l*NQ + w];
        const float gc = cosf(th), gs = sinf(th);
        float* Rr = S + ((cur*2 + 0)*DIM)*SIMK;
        float* Ri = S + ((cur*2 + 1)*DIM)*SIMK;
        for (int t = tid; t < (DIM/2) * SIMK; t += blockDim.x) {
          int p = t / SIMK, kk = t % SIMK;
          int j0 = ((p >> bpos) << (bpos + 1)) | (p & (mask - 1));
          int j1 = j0 | mask;
          float ar = Rr[j0*SIMK + kk], ai = Ri[j0*SIMK + kk];
          float br = Rr[j1*SIMK + kk], bi = Ri[j1*SIMK + kk];
          Rr[j0*SIMK + kk] = gc*ar + gs*bi;   // Re(gc*a - i*gs*b)
          Ri[j0*SIMK + kk] = gc*ai - gs*br;   // Im
          Rr[j1*SIMK + kk] = gs*ai + gc*br;   // Re(-i*gs*a + gc*b)
          Ri[j1*SIMK + kk] = gc*bi - gs*ar;   // Im
        }
        __syncthreads();
      }
      int nxt = cur ^ 1;
      for (int t = tid; t < DIM * SIMK; t += blockDim.x) {
        int j = t / SIMK, kk = t % SIMK;
        int pj = ring_perm(j);
        S[((nxt*2 + 0)*DIM + j)*SIMK + kk] = S[((cur*2 + 0)*DIM + pj)*SIMK + kk];
        S[((nxt*2 + 1)*DIM + j)*SIMK + kk] = S[((cur*2 + 1)*DIM + pj)*SIMK + kk];
      }
      __syncthreads();
      cur = nxt;
    }
    for (int t = tid; t < DIM * SIMK; t += blockDim.x) {
      int j = t / SIMK, kk = t % SIMK;
      Ur16[j*DIM + kbase + kk] = (_Float16)S[((cur*2 + 0)*DIM + j)*SIMK + kk];
      Ui16[j*DIM + kbase + kk] = (_Float16)S[((cur*2 + 1)*DIM + j)*SIMK + kk];
    }
  } else if (bid == SIM_BLOCKS) {
    // transposed f16 projection weights PW[16][2048], rows 8..15 = 0
    for (int t = tid; t < 16 * IN_F; t += blockDim.x) {
      int n = t / IN_F, k = t % IN_F;
      PW[t] = (_Float16)((n < NQ) ? proj_w[k*NQ + n] : 0.0f);
    }
  } else {
    // W2[c][j] = sum_w sign(j,w) * cls_w[w][c], padded to 1008 cols
    int nb = gridDim.x - (SIM_BLOCKS + 1);
    for (int t = (bid - (SIM_BLOCKS + 1)) * blockDim.x + tid;
         t < NCLS_PAD * DIM; t += nb * blockDim.x) {
      int c = t / DIM, j = t % DIM;
      float v = 0.0f;
      if (c < NCLS) {
#pragma unroll
        for (int w = 0; w < NQ; ++w) {
          float s = ((j >> (7 - w)) & 1) ? -1.0f : 1.0f;
          v += s * cls_w[w*NCLS + c];
        }
      }
      W2[t] = (_Float16)v;
    }
  }
}

__global__ void __launch_bounds__(WAVES_PER_BLOCK * 32)
pqc_main(const float* __restrict__ features,
         const float* __restrict__ proj_b,
         const float* __restrict__ cls_b,
         const char* __restrict__ ws,
         float* __restrict__ out) {
  extern __shared__ char smem[];
  const _Float16* Ur16 = (const _Float16*)(ws + WS_UR);
  const _Float16* Ui16 = (const _Float16*)(ws + WS_UI);
  const _Float16* W2   = (const _Float16*)(ws + WS_W2);
  const _Float16* PW   = (const _Float16*)(ws + WS_PW);

  const int lane   = threadIdx.x & 31;
  const int wave   = threadIdx.x >> 5;
  const int nIdx   = lane & 15;      // M row (A) / N col (B) owned by lane
  const int hiHalf = lane >> 4;      // 0: K 0..7,16..23   1: K 8..15,24..31
  const int koff   = hiHalf * 8;
  const int tileRow = (blockIdx.x * WAVES_PER_BLOCK + wave) * 16;

  _Float16* Xs = (_Float16*)(smem + wave * SMEM_PER_WAVE);          // [16][256]
  _Float16* Ps = Xs + 16 * DIM;                                     // [16][256]
  float*    cs = (float*)(smem + wave * SMEM_PER_WAVE + 16384);     // [16][8][2]

  // ---- Step 1: projection GEMM (M=16 rows, N=16 (8 used), K=2048), tanh -> cos/sin
  {
    v8f acc = {};
    const float*    fBase = features + (size_t)(tileRow + nIdx) * IN_F;
    const _Float16* bBase = PW + nIdx * IN_F;
    for (int k0 = 0; k0 < IN_F; k0 += 32) {
      v8f f0 = *(const v8f*)(fBase + k0 + koff);
      v8f f1 = *(const v8f*)(fBase + k0 + koff + 16);
      v16h a;
#pragma unroll
      for (int i = 0; i < 8; ++i) { a[i] = (_Float16)f0[i]; a[i + 8] = (_Float16)f1[i]; }
      v16h b = ldab(bBase + k0 + koff);
      acc = __builtin_amdgcn_wmma_f32_16x16x32_f16(false, a, false, b, (short)0, acc, false, false);
    }
    if (nIdx < NQ) {
      float pb = proj_b[nIdx];
#pragma unroll
      for (int v = 0; v < 8; ++v) {
        int m = v + hiHalf * 8;
        float ang = tanhf(acc[v] + pb) * 3.14159265358979f;
        cs[(m*NQ + nIdx)*2 + 0] = cosf(0.5f * ang);
        cs[(m*NQ + nIdx)*2 + 1] = sinf(0.5f * ang);
      }
    }
  }

  // ---- Step 2: build real product-state X[16][256] in LDS (f16)
  for (int r = 0; r < 16; ++r) {
    float cw[NQ], sw[NQ];
#pragma unroll
    for (int w = 0; w < NQ; ++w) { cw[w] = cs[(r*NQ + w)*2]; sw[w] = cs[(r*NQ + w)*2 + 1]; }
#pragma unroll
    for (int jj = 0; jj < 8; ++jj) {
      int j = lane * 8 + jj;
      float prod = 1.0f;
#pragma unroll
      for (int w = 0; w < NQ; ++w) prod *= ((j >> (7 - w)) & 1) ? sw[w] : cw[w];
      Xs[r*DIM + j] = (_Float16)prod;
    }
  }

  // ---- Step 3: Y = X @ U^T (re & im), probs = |Y|^2 -> LDS (f16)
  {
    // A-fragments (per-wave X rows) are N-tile invariant: cache all 8 K-steps in VGPRs
    v16h aF[8];
    {
      const _Float16* aBase = Xs + nIdx * DIM;
#pragma unroll
      for (int k = 0; k < 8; ++k) aF[k] = ldab(aBase + k * 32 + koff);
    }
    for (int jt = 0; jt < 16; ++jt) {
      v8f cr = {}, ci = {};
      const _Float16* brB = Ur16 + (size_t)(jt*16 + nIdx) * DIM;
      const _Float16* biB = Ui16 + (size_t)(jt*16 + nIdx) * DIM;
      if (jt + 1 < 16) {                       // warm next tile (global_prefetch_b8)
        __builtin_prefetch(brB + 16*DIM, 0, 3);
        __builtin_prefetch(biB + 16*DIM, 0, 3);
        __builtin_prefetch(brB + 16*DIM + 128, 0, 3);
        __builtin_prefetch(biB + 16*DIM + 128, 0, 3);
      }
#pragma unroll
      for (int k = 0; k < 8; ++k) {
        v16h br = ldab(brB + k * 32 + koff);
        v16h bi = ldab(biB + k * 32 + koff);
        cr = __builtin_amdgcn_wmma_f32_16x16x32_f16(false, aF[k], false, br, (short)0, cr, false, false);
        ci = __builtin_amdgcn_wmma_f32_16x16x32_f16(false, aF[k], false, bi, (short)0, ci, false, false);
      }
#pragma unroll
      for (int v = 0; v < 8; ++v) {
        int m = v + hiHalf * 8;
        float p = cr[v]*cr[v] + ci[v]*ci[v];
        Ps[m*DIM + jt*16 + nIdx] = (_Float16)p;
      }
    }
  }

  // ---- Step 4: Out = probs @ W2 + cls_b  (N padded to 1008)
  {
    v16h aP[8];
    {
      const _Float16* aBase = Ps + nIdx * DIM;
#pragma unroll
      for (int k = 0; k < 8; ++k) aP[k] = ldab(aBase + k * 32 + koff);
    }
    for (int ct = 0; ct < NCLS_PAD / 16; ++ct) {
      v8f acc = {};
      const _Float16* bB = W2 + (size_t)(ct*16 + nIdx) * DIM;
      if (ct + 1 < NCLS_PAD / 16) {
        __builtin_prefetch(bB + 16*DIM, 0, 3);
        __builtin_prefetch(bB + 16*DIM + 128, 0, 3);
      }
#pragma unroll
      for (int k = 0; k < 8; ++k) {
        v16h b = ldab(bB + k * 32 + koff);
        acc = __builtin_amdgcn_wmma_f32_16x16x32_f16(false, aP[k], false, b, (short)0, acc, false, false);
      }
      int ccol = ct * 16 + nIdx;
      if (ccol < NCLS) {
        float bias = cls_b[ccol];
#pragma unroll
        for (int v = 0; v < 8; ++v) {
          int m = v + hiHalf * 8;
          out[(size_t)(tileRow + m) * NCLS + ccol] = acc[v] + bias;
        }
      }
    }
  }
}

extern "C" void kernel_launch(void* const* d_in, const int* in_sizes, int n_in,
                              void* d_out, int out_size, void* d_ws, size_t ws_size,
                              hipStream_t stream) {
  (void)in_sizes; (void)n_in; (void)out_size; (void)ws_size;
  const float* features  = (const float*)d_in[0];
  const float* proj_w    = (const float*)d_in[1];
  const float* proj_b    = (const float*)d_in[2];
  const float* q_weights = (const float*)d_in[3];
  const float* cls_w     = (const float*)d_in[4];
  const float* cls_b     = (const float*)d_in[5];
  float* out = (float*)d_out;
  char*  ws  = (char*)d_ws;

  // 8 sim blocks (U columns) + 1 PW block + 11 W2 blocks
  pqc_precompute<<<20, 256, 2 * 2 * DIM * SIMK * (int)sizeof(float), stream>>>(
      proj_w, q_weights, cls_w, ws);

  int blocks = BATCH / (WAVES_PER_BLOCK * 16);   // 256
  pqc_main<<<blocks, WAVES_PER_BLOCK * 32, WAVES_PER_BLOCK * SMEM_PER_WAVE, stream>>>(
      features, proj_b, cls_b, ws, out);
}
